// CausalAttention_50706383897047
// MI455X (gfx1250) — matrix operand order repaired and checked by reading, not compiled, not run on hardware
//
#include <hip/hip_runtime.h>
#include <stdint.h>

#define BATCH 8
#define SEQ   1024
#define CH    768
#define NH    12
#define HD    64
#define C3    2304

typedef __bf16 bf16_t;
typedef __attribute__((ext_vector_type(16))) __bf16 v16bf;
typedef __attribute__((ext_vector_type(8)))  float  v8f;
typedef __attribute__((ext_vector_type(4)))  uint32_t u32x4;
typedef __attribute__((ext_vector_type(8)))  int32_t  i32x8;
typedef __attribute__((ext_vector_type(4)))  int32_t  i32x4;

union FragAB {
  v16bf v;
  uint4 q[2];
};
union Pack8 {
  uint4 q;
  uint16_t h[8];
};

static __device__ __forceinline__ uint16_t f2bf_bits(float f) {
  uint32_t u = __builtin_bit_cast(uint32_t, f);
  uint32_t r = u + 0x7FFFu + ((u >> 16) & 1u);   // round-to-nearest-even
  return (uint16_t)(r >> 16);
}
static __device__ __forceinline__ uint32_t pack2(float a, float b) {
  return (uint32_t)f2bf_bits(a) | ((uint32_t)f2bf_bits(b) << 16);
}
static __device__ __forceinline__ v8f vzero() {
  v8f z;
#pragma unroll
  for (int i = 0; i < 8; ++i) z[i] = 0.0f;
  return z;
}
static __device__ __forceinline__ v8f wmma_bf16(const FragAB& a, const FragAB& b, v8f c) {
  return __builtin_amdgcn_wmma_f32_16x16x32_bf16(false, a.v, false, b.v, (short)0, c,
                                                 false, false);
}

// LDS byte offset: generic LDS addresses truncate to addr[31:0] (ISA 10.2).
static __device__ __forceinline__ uint32_t lds_addr_of(const void* p) {
  return (uint32_t)(uintptr_t)p;
}

// 2-D TDM tile load of 2-byte elements, global -> LDS, with per-row LDS padding
// so the LDS layout keeps our bank-conflict padding. D# per ISA 8.3/8.4:
//   group0: count=1 | lds_addr | global_addr(57b) | type=2
//   group1: data_size=2B, pad_enable, pad_interval/amount, dims, strides
// groups 2/3 (and trailing group) zero: 2-D tensor. Tracked by TENSORcnt.
// This toolchain exposes the 6-arg builtin (extra int32x8 group before cpol).
static __device__ __forceinline__ void tdm_load_2d_bf16(
    uint32_t lds_byte, uint64_t gaddr, uint32_t cols, uint32_t rows,
    uint32_t gstride_elems, uint32_t pad_interval_code, uint32_t pad_amount_code) {
  u32x4 g0;
  g0.x = 1u;                                         // count=1 (valid user D#)
  g0.y = lds_byte;                                   // lds_addr
  g0.z = (uint32_t)gaddr;                            // global_addr[31:0]
  g0.w = (uint32_t)(gaddr >> 32) | (2u << 30);       // global_addr[56:32] | type=2
  i32x8 g1;
  g1[0] = (int)((1u << 16) |                         // data_size = 2 bytes
                (1u << 20) |                         // pad_enable
                (pad_interval_code << 22) |
                (pad_amount_code << 25));
  g1[1] = (int)(cols << 16);                         // tensor_dim0[15:0]
  g1[2] = (int)(rows << 16);                         // tensor_dim1[15:0]
  g1[3] = (int)(cols << 16);                         // tile_dim0
  g1[4] = (int)rows;                                 // tile_dim1 (tile_dim2=0)
  g1[5] = (int)gstride_elems;                        // tensor_dim0_stride[31:0]
  g1[6] = 0;
  g1[7] = 0;
  i32x4 z4;
#pragma unroll
  for (int i = 0; i < 4; ++i) z4[i] = 0;
  i32x8 z8;
#pragma unroll
  for (int i = 0; i < 8; ++i) z8[i] = 0;
  __builtin_amdgcn_tensor_load_to_lds(g0, g1, z4, z4, z8, 0);
}

// ---------------------------------------------------------------------------
// GEMM1: qkv[8192,2304] = x[8192,768] @ W_in[768,2304] + b_in  -> bf16
// 128x128x32 block tiles, 256 threads = 8 waves, 16x128 strip per wave.
// (f32 inputs -> bf16 conversion happens while staging, so no TDM here.)
// ---------------------------------------------------------------------------
#define BM 128
#define BN 128
#define BK 32
#define LDT 40   // BK + 8 pad (u16 elements); 80-byte row stride, 16B aligned

__global__ __launch_bounds__(256) void gemm_qkv_kernel(
    const float* __restrict__ x, const float* __restrict__ Win,
    const float* __restrict__ bin, uint16_t* __restrict__ qkv) {
  __shared__ __align__(16) uint16_t As[BM * LDT];
  __shared__ __align__(16) uint16_t Bs[BN * LDT];
  const int tid  = threadIdx.x;
  const int n0   = blockIdx.x * BN;
  const int m0   = blockIdx.y * BM;
  const int wave = tid >> 5, lane = tid & 31;
  const int lrow = lane & 15, lhi = lane >> 4;

  v8f acc[8];
#pragma unroll
  for (int nt = 0; nt < 8; ++nt) acc[nt] = vzero();

  for (int kb = 0; kb < CH; kb += BK) {
    const float* xa = x + (size_t)m0 * CH + kb;
#pragma unroll
    for (int p = 0; p < 4; ++p) {
      int r = (tid >> 3) + p * 32;
      int c = (tid & 7) * 4;
      float4 f = *(const float4*)(xa + (size_t)r * CH + c);
      uint2 u; u.x = pack2(f.x, f.y); u.y = pack2(f.z, f.w);
      *(uint2*)(&As[r * LDT + c]) = u;
    }
    const float* wb = Win + (size_t)kb * C3 + n0;
#pragma unroll
    for (int p = 0; p < 4; ++p) {
      int k = (tid >> 5) + p * 8;
      int n = (tid & 31) * 4;
      float4 f = *(const float4*)(wb + (size_t)k * C3 + n);
      Bs[(n + 0) * LDT + k] = f2bf_bits(f.x);
      Bs[(n + 1) * LDT + k] = f2bf_bits(f.y);
      Bs[(n + 2) * LDT + k] = f2bf_bits(f.z);
      Bs[(n + 3) * LDT + k] = f2bf_bits(f.w);
    }
    if (kb + BK < CH) {   // prefetch next K-tile -> global_prefetch_b8
      __builtin_prefetch(xa + (size_t)(tid >> 3) * CH + BK, 0, 1);
      __builtin_prefetch(wb + (size_t)(BK + (tid >> 5)) * C3, 0, 1);
    }
    __syncthreads();

    FragAB a;
    const uint16_t* ap = &As[(wave * 16 + lrow) * LDT + lhi * 8];
    a.q[0] = *(const uint4*)(ap);
    a.q[1] = *(const uint4*)(ap + 16);
#pragma unroll
    for (int nt = 0; nt < 8; ++nt) {
      FragAB b;
      const uint16_t* bp = &Bs[(nt * 16 + lrow) * LDT + lhi * 8];
      b.q[0] = *(const uint4*)(bp);
      b.q[1] = *(const uint4*)(bp + 16);
      acc[nt] = wmma_bf16(a, b, acc[nt]);
    }
    __syncthreads();
  }
#pragma unroll
  for (int nt = 0; nt < 8; ++nt) {
    int col = n0 + nt * 16 + lrow;
    float bias = bin[col];
#pragma unroll
    for (int i = 0; i < 8; ++i) {
      int row = m0 + wave * 16 + lhi * 8 + i;
      qkv[(size_t)row * C3 + col] = f2bf_bits(acc[nt][i] + bias);
    }
  }
}

// ---------------------------------------------------------------------------
// Flash-style causal attention. Grid: (T/64, B*H), 128 threads = 4 waves.
// qkv row layout: [k(768) | q(768) | v(768)] per (b,t) row (k first!).
// Q and K tiles are straight bf16 copies -> Tensor Data Mover (TDM) loads
// with LDS row padding (tile row 128B = 32 DW, pad 4 DW -> 144B stride).
// ---------------------------------------------------------------------------
__global__ __launch_bounds__(128) void attn_kernel(
    const uint16_t* __restrict__ qkv, uint16_t* __restrict__ ybf) {
  __shared__ __align__(16) uint16_t Qs[64 * 72];
  __shared__ __align__(16) uint16_t Ks[64 * 72];
  __shared__ __align__(16) uint16_t Vs[64 * 72];   // transposed: [d][key]
  __shared__ __align__(16) uint16_t Ps[64 * 72];
  const int tid  = threadIdx.x;
  const int qb   = blockIdx.x;           // query block 0..15
  const int bh   = blockIdx.y;           // 0..95
  const int bb   = bh / NH, h = bh % NH;
  const int q0   = qb * 64;
  const int wave = tid >> 5, lane = tid & 31;
  const int lrow = lane & 15, lhi = lane >> 4;
  const size_t rowbase = (size_t)bb * SEQ;

  if (wave == 0) {                       // TDM: Q block 64x64 bf16 -> LDS
    tdm_load_2d_bf16(lds_addr_of(Qs),
                     (uint64_t)(uintptr_t)(qkv + (rowbase + q0) * C3 + CH + h * HD),
                     64, 64, C3, /*interval 32DW*/ 4, /*pad 4DW*/ 3);
    __builtin_amdgcn_s_wait_tensorcnt(0);
  }
  __syncthreads();

  FragAB qf[2];
#pragma unroll
  for (int kt = 0; kt < 2; ++kt) {
    const uint16_t* p = &Qs[(wave * 16 + lrow) * 72 + kt * 32 + lhi * 8];
    qf[kt].q[0] = *(const uint4*)p;
    qf[kt].q[1] = *(const uint4*)(p + 16);
  }

  float mrun[8], lrun[8];
  v8f yacc[4];
#pragma unroll
  for (int i = 0; i < 8; ++i) { mrun[i] = -3.0e38f; lrun[i] = 0.0f; }
#pragma unroll
  for (int dt = 0; dt < 4; ++dt) yacc[dt] = vzero();

  for (int kblk = 0; kblk <= qb; ++kblk) {
    __syncthreads();                     // Ks/Vs no longer read by any wave
    if (wave == 0) {                     // TDM: K block 64x64 bf16 -> LDS
      tdm_load_2d_bf16(lds_addr_of(Ks),
                       (uint64_t)(uintptr_t)(qkv + (rowbase + kblk * 64) * C3 + h * HD),
                       64, 64, C3, 4, 3);
    }
#pragma unroll
    for (int p = 0; p < 4; ++p) {        // V needs transpose -> manual staging
      int idx = tid + p * 128;
      int r = idx >> 3;
      int c = (idx & 7) * 8;
      Pack8 pv;
      pv.q = *(const uint4*)(qkv + (rowbase + kblk * 64 + r) * C3 + 2 * CH + h * HD + c);
#pragma unroll
      for (int j = 0; j < 8; ++j) Vs[(c + j) * 72 + r] = pv.h[j];
    }
    if (wave == 0) __builtin_amdgcn_s_wait_tensorcnt(0);
    __syncthreads();

    v8f sc[4];                           // scores 16x64 per wave
#pragma unroll
    for (int nt = 0; nt < 4; ++nt) sc[nt] = vzero();
#pragma unroll
    for (int kt = 0; kt < 2; ++kt) {
#pragma unroll
      for (int nt = 0; nt < 4; ++nt) {
        FragAB b;                        // B^t[key][d] == K rows as stored
        const uint16_t* bp = &Ks[(nt * 16 + lrow) * 72 + kt * 32 + lhi * 8];
        b.q[0] = *(const uint4*)bp;
        b.q[1] = *(const uint4*)(bp + 16);
        sc[nt] = wmma_bf16(qf[kt], b, sc[nt]);
      }
    }
    if (kblk == qb) {                    // causal mask on diagonal block
#pragma unroll
      for (int nt = 0; nt < 4; ++nt) {
        int col = nt * 16 + lrow;
#pragma unroll
        for (int i = 0; i < 8; ++i) {
          int row = wave * 16 + lhi * 8 + i;
          if (col > row) sc[nt][i] = -3.0e38f;
        }
      }
    }
    // online softmax; row m lives in VGPR (m%8), lanes (m<8 ? 0-15 : 16-31)
#pragma unroll
    for (int i = 0; i < 8; ++i) {
      float v = sc[0][i];
      v = fmaxf(v, sc[1][i]); v = fmaxf(v, sc[2][i]); v = fmaxf(v, sc[3][i]);
#pragma unroll
      for (int off = 1; off < 16; off <<= 1) v = fmaxf(v, __shfl_xor(v, off, 32));
      float mnew  = fmaxf(mrun[i], v);
      float scale = __expf(mrun[i] - mnew);
      mrun[i] = mnew;
      float s = 0.0f;
#pragma unroll
      for (int nt = 0; nt < 4; ++nt) {
        float pe = __expf(sc[nt][i] - mnew);
        sc[nt][i] = pe;
        s += pe;
      }
#pragma unroll
      for (int off = 1; off < 16; off <<= 1) s += __shfl_xor(s, off, 32);
      lrun[i] = lrun[i] * scale + s;
#pragma unroll
      for (int dt = 0; dt < 4; ++dt) yacc[dt][i] *= scale;
    }
    // P -> LDS (wave-private rows) to reshape into A-fragments
#pragma unroll
    for (int nt = 0; nt < 4; ++nt)
#pragma unroll
      for (int i = 0; i < 8; ++i)
        Ps[(wave * 16 + lhi * 8 + i) * 72 + nt * 16 + lrow] = f2bf_bits(sc[nt][i]);
    // y += P @ V  (B^t[d][key] == Vs as stored)
#pragma unroll
    for (int kt = 0; kt < 2; ++kt) {
      FragAB a;
      const uint16_t* ap = &Ps[(wave * 16 + lrow) * 72 + kt * 32 + lhi * 8];
      a.q[0] = *(const uint4*)ap;
      a.q[1] = *(const uint4*)(ap + 16);
#pragma unroll
      for (int dt = 0; dt < 4; ++dt) {
        FragAB b;
        const uint16_t* bp = &Vs[(dt * 16 + lrow) * 72 + kt * 32 + lhi * 8];
        b.q[0] = *(const uint4*)bp;
        b.q[1] = *(const uint4*)(bp + 16);
        yacc[dt] = wmma_bf16(a, b, yacc[dt]);
      }
    }
  }
#pragma unroll
  for (int i = 0; i < 8; ++i) {
    float inv = 1.0f / lrun[i];
    size_t grow = rowbase + q0 + wave * 16 + lhi * 8 + i;
#pragma unroll
    for (int dt = 0; dt < 4; ++dt) {
      int col = h * HD + dt * 16 + lrow;
      ybf[grow * CH + col] = f2bf_bits(yacc[dt][i] * inv);
    }
  }
}

// ---------------------------------------------------------------------------
// GEMM2: out[8192,768] = y_bf16 @ W_out[768,768] + b_out -> f32
// A tile is a straight bf16 copy -> TDM load (row 64B = 16 DW, pad 4 DW).
// ---------------------------------------------------------------------------
__global__ __launch_bounds__(256) void gemm_out_kernel(
    const uint16_t* __restrict__ ybf, const float* __restrict__ Wout,
    const float* __restrict__ bout, float* __restrict__ out) {
  __shared__ __align__(16) uint16_t As[BM * LDT];
  __shared__ __align__(16) uint16_t Bs[BN * LDT];
  const int tid  = threadIdx.x;
  const int n0   = blockIdx.x * BN;
  const int m0   = blockIdx.y * BM;
  const int wave = tid >> 5, lane = tid & 31;
  const int lrow = lane & 15, lhi = lane >> 4;

  v8f acc[8];
#pragma unroll
  for (int nt = 0; nt < 8; ++nt) acc[nt] = vzero();

  for (int kb = 0; kb < CH; kb += BK) {
    if (wave == 0) {                     // TDM: A tile 128x32 bf16 -> LDS
      tdm_load_2d_bf16(lds_addr_of(As),
                       (uint64_t)(uintptr_t)(ybf + (size_t)m0 * CH + kb),
                       BK, BM, CH, /*interval 16DW*/ 3, /*pad 4DW*/ 3);
    }
    const float* wb = Wout + (size_t)kb * CH + n0;
#pragma unroll
    for (int p = 0; p < 4; ++p) {
      int k = (tid >> 5) + p * 8;
      int n = (tid & 31) * 4;
      float4 f = *(const float4*)(wb + (size_t)k * CH + n);
      Bs[(n + 0) * LDT + k] = f2bf_bits(f.x);
      Bs[(n + 1) * LDT + k] = f2bf_bits(f.y);
      Bs[(n + 2) * LDT + k] = f2bf_bits(f.z);
      Bs[(n + 3) * LDT + k] = f2bf_bits(f.w);
    }
    if (kb + BK < CH)
      __builtin_prefetch(wb + (size_t)(BK + (tid >> 5)) * CH, 0, 1);
    if (wave == 0) __builtin_amdgcn_s_wait_tensorcnt(0);
    __syncthreads();

    FragAB a;
    const uint16_t* ap = &As[(wave * 16 + lrow) * LDT + lhi * 8];
    a.q[0] = *(const uint4*)(ap);
    a.q[1] = *(const uint4*)(ap + 16);
#pragma unroll
    for (int nt = 0; nt < 8; ++nt) {
      FragAB b;
      const uint16_t* bp = &Bs[(nt * 16 + lrow) * LDT + lhi * 8];
      b.q[0] = *(const uint4*)(bp);
      b.q[1] = *(const uint4*)(bp + 16);
      acc[nt] = wmma_bf16(a, b, acc[nt]);
    }
    __syncthreads();
  }
#pragma unroll
  for (int nt = 0; nt < 8; ++nt) {
    int col = n0 + nt * 16 + lrow;
    float bias = bout[col];
#pragma unroll
    for (int i = 0; i < 8; ++i) {
      int row = m0 + wave * 16 + lhi * 8 + i;
      out[(size_t)row * CH + col] = acc[nt][i] + bias;
    }
  }
}

// ---------------------------------------------------------------------------
extern "C" void kernel_launch(void* const* d_in, const int* in_sizes, int n_in,
                              void* d_out, int out_size, void* d_ws, size_t ws_size,
                              hipStream_t stream) {
  (void)in_sizes; (void)n_in; (void)out_size; (void)ws_size;
  const float* x    = (const float*)d_in[0];
  const float* Win  = (const float*)d_in[1];
  const float* bin  = (const float*)d_in[2];
  const float* Wout = (const float*)d_in[3];
  const float* bout = (const float*)d_in[4];
  float* out = (float*)d_out;

  uint16_t* qkv = (uint16_t*)d_ws;                       // 8192*2304 bf16
  uint16_t* ybf = qkv + (size_t)BATCH * SEQ * C3;        // 8192*768  bf16

  gemm_qkv_kernel<<<dim3(C3 / BN, (BATCH * SEQ) / BM), 256, 0, stream>>>(
      x, Win, bin, qkv);
  attn_kernel<<<dim3(SEQ / 64, BATCH * NH), 128, 0, stream>>>(qkv, ybf);
  gemm_out_kernel<<<dim3(CH / BN, (BATCH * SEQ) / BM), 256, 0, stream>>>(
      ybf, Wout, bout, out);
}
